// ViT_5506148074198
// MI455X (gfx1250) — matrix operand order repaired and checked by reading, not compiled, not run on hardware
//
#include <hip/hip_runtime.h>
#include <hip/hip_bf16.h>

// ============================================================================
// ViT-Base forward for MI455X (gfx1250, wave32, WMMA + async-LDS).
//  - All GEMMs: v_wmma_f32_16x16x32_bf16, f32 accumulate.
//  - Weights pre-converted once per launch to bf16 in fragment-swizzled
//    global layout -> B tiles are contiguous 1KB blocks, loaded with
//    GLOBAL_LOAD_ASYNC_TO_LDS_B128 (ASYNCcnt, double-buffered).
//  - Activations are produced in bf16 row-major; 16B row chunks map to
//    contiguous 16B chunks of the ISA A-fragment layout, so A tiles also
//    go through the async path with zero VALU swizzle work.
//  - Attention is fused (flash-style online softmax), bf16 in/out.
// Workspace: ~242 MB (f32 x + bf16 activations + 135 MB swizzled weights).
// ============================================================================

typedef __attribute__((ext_vector_type(16))) __bf16 v16bf;
typedef __attribute__((ext_vector_type(8)))  float  v8f;
typedef int v4i_ __attribute__((__vector_size__(16)));   // builtin param type

#define BATCH   32
#define SEQ     197
#define DIM     768
#define HEADS   12
#define HDIM    64
#define LAYERS  12
#define FFDIM   3072
#define NCLS    1000
#define NPATCH  196
#define M_TOK   (BATCH * SEQ)   // 6304

#ifndef __has_builtin
#define __has_builtin(x) 0
#endif
#if __has_builtin(__builtin_amdgcn_global_load_async_to_lds_b128)
#define HAS_ASYNC_LDS 1
#else
#define HAS_ASYNC_LDS 0
#endif

// ---- 16-bit A-matrix 16x32 fragment mapping (ISA 7.12.2) -------------------
__device__ __forceinline__ int fragA_half(int k) { return (k >> 3) & 1; }
__device__ __forceinline__ int fragA_slot(int k) { return (k & 7) | ((k & 16) >> 1); }
__device__ __forceinline__ int fragA_k(int half, int s) {
    return (s & 7) + (half << 3) + ((s >> 3) << 4);
}
// ---- 16-bit B-matrix 32x16 fragment mapping (SWMMAC dense-B table) ---------
__device__ __forceinline__ int fragB_k(int half, int s) { return (s & 7) + (half << 3) + ((s >> 3) << 4); }

__device__ __forceinline__ float gelu_exact(float x) {
    return 0.5f * x * (1.f + erff(x * 0.70710678118654752f));
}

// 16-byte global -> LDS copy. Async (ASYNCcnt) when the gfx1250 builtin is
// available; synchronous uint4 copy otherwise (same layout, same correctness).
__device__ __forceinline__ void copy16_g2l(const void* g, void* l) {
#if HAS_ASYNC_LDS
    __builtin_amdgcn_global_load_async_to_lds_b128((v4i_*)g, (v4i_*)l, 0, 0);
#else
    *(uint4*)l = *(const uint4*)g;
#endif
}

template <int N>
__device__ __forceinline__ void async_wait() {
#if HAS_ASYNC_LDS
#if __has_builtin(__builtin_amdgcn_s_wait_asynccnt)
    __builtin_amdgcn_s_wait_asynccnt(N);
#else
    asm volatile("s_wait_asynccnt %0" :: "i"(N) : "memory");
#endif
#endif
}

// ============================================================================
// Weight pre-swizzle: f32 [K,N] row-major -> bf16 fragment-block layout.
// Block (kt, nt) = 512 bf16 = [32 lanes][16 slots]; element (k,n):
// lane = ((k>>3)&1)*16 + (n%16), slot = (k&7) | ((k&16)>>1). N padded to N64 (zeros).
// Out index = ((z*(K/32)+kt)*(N64/16)+nt)*512 + lane*16 + slot.
// ============================================================================
__global__ void weight_swizzle_kernel(const float* __restrict__ Win,
                                      __bf16* __restrict__ Wout,
                                      int K, int N, int N64, long inStrideZ,
                                      long total)
{
    long idx = (long)blockIdx.x * 256 + threadIdx.x;
    if (idx >= total) return;
    int  slot = (int)(idx & 15);
    int  lane = (int)((idx >> 4) & 31);
    long blk  = idx >> 9;
    int  nBlocks = N64 >> 4;
    int  KT      = K >> 5;
    int  nt = (int)(blk % nBlocks);
    long t2 = blk / nBlocks;
    int  kt = (int)(t2 % KT);
    int  z  = (int)(t2 / KT);
    int  k = kt * 32 + fragB_k(lane >> 4, slot);
    int  n = nt * 16 + (lane & 15);
    float v = (n < N) ? Win[z * inStrideZ + (long)k * N + n] : 0.f;
    Wout[idx] = (__bf16)v;
}

// ============================================================================
// WMMA GEMM: C = act(A_bf16 @ Wswz + bias) [+Res].  Block tile 128x64, K-step
// 32, double-buffered LDS fed by async b128 copies (2 A + 1 B per thread per
// stage -> exactly 3 async ops per wave per stage; s_wait_asynccnt 3 overlaps
// next-stage loads with current-stage WMMAs).
// ============================================================================
template <int ACT, int HAS_RES, int OUT_BF16>
__global__ __launch_bounds__(256) void gemm_wmma_kernel(
    const __bf16* __restrict__ A, int lda, long aOffZ,
    const __bf16* __restrict__ Wsw, long wOffZ, int nBlocks,
    const float* __restrict__ bias, long bOffZ,
    void* __restrict__ Cout, int ldc, long cOffZ,
    const float* __restrict__ Res,
    int M, int N, int K)
{
    __shared__ alignas(32) __bf16 ldsA[2][8][32][16];   // 2 x 8KB
    __shared__ alignas(32) __bf16 ldsB[2][4][32][16];   // 2 x 4KB

    const int z = blockIdx.z;
    A   += (long)z * aOffZ;
    Wsw += (long)z * wOffZ;
    bias += (long)z * bOffZ;

    const int row0 = blockIdx.y * 128;
    const int col0 = blockIdx.x * 64;
    const int tid  = threadIdx.x;
    const int lane = tid & 31;
    const int wave = tid >> 5;
    const int ntBase = col0 >> 4;

    v8f acc[4];
#pragma unroll
    for (int j = 0; j < 4; ++j)
#pragma unroll
        for (int r = 0; r < 8; ++r) acc[j][r] = 0.f;

    // Stage copy: A row chunks (16B) land on contiguous A-fragment chunks;
    // B fragments are contiguous 1KB blocks in the swizzled weight buffer.
    auto issue = [&](int kt, int buf) {
#pragma unroll
        for (int c0 = 0; c0 < 2; ++c0) {
            int c = tid + c0 * 256;          // 512 chunks: 128 rows x 4 chunks
            int m = c >> 2, j = c & 3;
            int r = row0 + m;                // may over-read into ws slack
            const __bf16* g = A + (long)r * lda + kt * 32 + j * 8;
            __bf16* l = &ldsA[buf][m >> 4][((j & 1) << 4) + (m & 15)][(j >> 1) * 8];
            copy16_g2l(g, l);
        }
        {
            int j = tid >> 6, cc = tid & 63; // 4 fragments x 64 chunks
            const __bf16* g = Wsw + ((long)kt * nBlocks + ntBase + j) * 512 + cc * 8;
            __bf16* l = &ldsB[buf][j][cc >> 1][(cc & 1) * 8];
            copy16_g2l(g, l);
        }
    };

    const int T = K >> 5;
    issue(0, 0);
    for (int t = 0; t < T; ++t) {
        const int cur = t & 1;
        if (t + 1 < T) { issue(t + 1, cur ^ 1); async_wait<3>(); }
        else           { async_wait<0>(); }
        __syncthreads();

        v16bf a = *(const v16bf*)&ldsA[cur][wave][lane][0];
#pragma unroll
        for (int j = 0; j < 4; ++j) {
            v16bf b = *(const v16bf*)&ldsB[cur][j][lane][0];
            acc[j] = __builtin_amdgcn_wmma_f32_16x16x32_bf16(
                false, a, false, b, (short)0, acc[j], false, false);
        }
        __syncthreads();
    }

    // Epilogue: C/D layout — lane holds rows m=(lane/16)*8+r, col = lane%16
    const int mBase = row0 + wave * 16 + ((lane >> 4) << 3);
    const int nInd  = lane & 15;
#pragma unroll
    for (int j = 0; j < 4; ++j) {
        int c = col0 + j * 16 + nInd;
        if (c < N) {
            float bv = bias[c];
#pragma unroll
            for (int r = 0; r < 8; ++r) {
                int m = mBase + r;
                if (m < M) {
                    float val = acc[j][r] + bv;
                    if (ACT == 1) val = gelu_exact(val);
                    if (OUT_BF16) {
                        ((__bf16*)Cout)[(long)z * cOffZ + (long)m * ldc + c] = (__bf16)val;
                    } else {
                        if (HAS_RES)
                            val += Res[(long)z * cOffZ + (long)m * ldc + c];
                        ((float*)Cout)[(long)z * cOffZ + (long)m * ldc + c] = val;
                    }
                }
            }
        }
    }
}

// ============================================================================
// Fused attention (bf16 q/k/v/o): one block per (b,h); 8 waves; wave w handles
// query tiles w and w+8. Key chunks of 32: QK^T (4 WMMAs), online softmax,
// P -> A-fragment via per-wave LDS re-swizzle, P@V (4 WMMAs).
// ============================================================================
__global__ __launch_bounds__(256) void attention_kernel(
    const __bf16* __restrict__ q, const __bf16* __restrict__ k,
    const __bf16* __restrict__ v, __bf16* __restrict__ o)
{
    __shared__ alignas(32) __bf16 ldsK[2][2][32][16];
    __shared__ alignas(32) __bf16 ldsV[4][32][16];
    __shared__ alignas(32) __bf16 ldsP[8][32][16];

    const int bh = blockIdx.x;
    const int b  = bh / HEADS;
    const int h  = bh % HEADS;
    const long base = (long)b * SEQ * DIM + (long)h * HDIM;
    const __bf16* qb = q + base;
    const __bf16* kb = k + base;
    const __bf16* vb = v + base;
    __bf16*       ob = o + base;

    const int tid  = threadIdx.x;
    const int lane = tid & 31;
    const int wave = tid >> 5;
    const int half = lane >> 4;
    const int l16  = lane & 15;
    const float scale = 0.125f;               // 1/sqrt(64)

    const int nQT    = (SEQ + 15) / 16;       // 13
    const int nChunk = (SEQ + 31) / 32;       // 7

    for (int qt = wave; qt < 16; qt += 8) {   // uniform 2 iterations per wave
        const int qRow0 = qt * 16;

        v16bf qf0, qf1;
#pragma unroll
        for (int s = 0; s < 16; ++s) {
            int kk  = fragA_k(half, s);
            int row = qRow0 + l16;
            __bf16 f0 = (__bf16)0.f, f1 = (__bf16)0.f;
            if (row < SEQ) {
                f0 = qb[(long)row * DIM + kk];
                f1 = qb[(long)row * DIM + 32 + kk];
            }
            qf0[s] = f0;
            qf1[s] = f1;
        }

        float mstat[8], lstat[8];
        v8f oacc[4];
#pragma unroll
        for (int r = 0; r < 8; ++r) { mstat[r] = -1e30f; lstat[r] = 0.f; }
#pragma unroll
        for (int nt = 0; nt < 4; ++nt)
#pragma unroll
            for (int r = 0; r < 8; ++r) oacc[nt][r] = 0.f;

        for (int ch = 0; ch < nChunk; ++ch) {
            const int key0 = ch * 32;
            __syncthreads();
            for (int i = tid; i < 2048; i += 256) {      // K^T fragments
                int s  = i & 15;
                int ln = (i >> 4) & 31;
                int f  = (i >> 9) & 1;
                int t  = i >> 10;
                int d  = f * 32 + fragB_k(ln >> 4, s);
                int keyRow = key0 + t * 16 + (ln & 15);
                ldsK[t][f][ln][s] =
                    (keyRow < SEQ) ? kb[(long)keyRow * DIM + d] : (__bf16)0.f;
            }
            for (int i = tid; i < 2048; i += 256) {      // V fragments
                int s  = i & 15;
                int ln = (i >> 4) & 31;
                int nt = i >> 9;
                int j  = fragB_k(ln >> 4, s);
                int keyRow = key0 + j;
                ldsV[nt][ln][s] = (keyRow < SEQ)
                    ? vb[(long)keyRow * DIM + nt * 16 + (ln & 15)] : (__bf16)0.f;
            }
            __syncthreads();

            v8f sc[2];
#pragma unroll
            for (int t = 0; t < 2; ++t) {
                v16bf kf0 = *(const v16bf*)&ldsK[t][0][lane][0];
                v16bf kf1 = *(const v16bf*)&ldsK[t][1][lane][0];
                v8f zz = {0.f, 0.f, 0.f, 0.f, 0.f, 0.f, 0.f, 0.f};
                zz = __builtin_amdgcn_wmma_f32_16x16x32_bf16(
                    false, qf0, false, kf0, (short)0, zz, false, false);
                zz = __builtin_amdgcn_wmma_f32_16x16x32_bf16(
                    false, qf1, false, kf1, (short)0, zz, false, false);
                sc[t] = zz;
            }

            float chmax[8];
#pragma unroll
            for (int r = 0; r < 8; ++r) chmax[r] = -1e30f;
#pragma unroll
            for (int t = 0; t < 2; ++t) {
                int key = key0 + t * 16 + l16;
#pragma unroll
                for (int r = 0; r < 8; ++r) {
                    float sv = sc[t][r] * scale;
                    if (key >= SEQ) sv = -1e30f;
                    sc[t][r] = sv;
                    chmax[r] = fmaxf(chmax[r], sv);
                }
            }
#pragma unroll
            for (int msk = 1; msk < 16; msk <<= 1)
#pragma unroll
                for (int r = 0; r < 8; ++r)
                    chmax[r] = fmaxf(chmax[r], __shfl_xor(chmax[r], msk, 32));

            float rescl[8], rsum[8];
#pragma unroll
            for (int r = 0; r < 8; ++r) {
                float nm = fmaxf(mstat[r], chmax[r]);
                rescl[r] = __expf(mstat[r] - nm);
                mstat[r] = nm;
                rsum[r]  = 0.f;
            }
#pragma unroll
            for (int t = 0; t < 2; ++t)
#pragma unroll
                for (int r = 0; r < 8; ++r) {
                    float p = __expf(sc[t][r] - mstat[r]);
                    sc[t][r] = p;
                    rsum[r] += p;
                }
#pragma unroll
            for (int msk = 1; msk < 16; msk <<= 1)
#pragma unroll
                for (int r = 0; r < 8; ++r)
                    rsum[r] += __shfl_xor(rsum[r], msk, 32);
#pragma unroll
            for (int r = 0; r < 8; ++r)
                lstat[r] = lstat[r] * rescl[r] + rsum[r];
#pragma unroll
            for (int nt = 0; nt < 4; ++nt)
#pragma unroll
                for (int r = 0; r < 8; ++r) oacc[nt][r] *= rescl[r];

            // P -> bf16 A-fragment via per-wave LDS (same-wave DS is in-order)
#pragma unroll
            for (int t = 0; t < 2; ++t)
#pragma unroll
                for (int r = 0; r < 8; ++r) {
                    int mm = half * 8 + r;
                    int jj = t * 16 + l16;
                    ldsP[wave][(fragA_half(jj) << 4) | mm][fragA_slot(jj)] =
                        (__bf16)sc[t][r];
                }
            v16bf pf = *(const v16bf*)&ldsP[wave][lane][0];
#pragma unroll
            for (int nt = 0; nt < 4; ++nt) {
                v16bf vf = *(const v16bf*)&ldsV[nt][lane][0];
                oacc[nt] = __builtin_amdgcn_wmma_f32_16x16x32_bf16(
                    false, pf, false, vf, (short)0, oacc[nt], false, false);
            }
        }

        if (qt < nQT) {
#pragma unroll
            for (int nt = 0; nt < 4; ++nt)
#pragma unroll
                for (int r = 0; r < 8; ++r) {
                    int row = qRow0 + half * 8 + r;
                    if (row < SEQ)
                        ob[(long)row * DIM + nt * 16 + l16] =
                            (__bf16)(oacc[nt][r] / lstat[r]);
                }
        }
    }
}

// ============================================================================
// LayerNorm: f32 in, bf16 out (feeds the async bf16 GEMM A path).
// ============================================================================
__global__ __launch_bounds__(256) void layernorm_kernel(
    const float* __restrict__ xin, const float* __restrict__ g,
    const float* __restrict__ bb, __bf16* __restrict__ out, int D)
{
    const long row = blockIdx.x;
    const float* xr = xin + row * D;
    float s1 = 0.f, s2 = 0.f;
    for (int j = threadIdx.x; j < D; j += 256) {
        float xv = xr[j];
        s1 += xv; s2 += xv * xv;
    }
    for (int m = 16; m >= 1; m >>= 1) {
        s1 += __shfl_xor(s1, m, 32);
        s2 += __shfl_xor(s2, m, 32);
    }
    __shared__ float sh1[8], sh2[8], stats[2];
    const int lane = threadIdx.x & 31, wave = threadIdx.x >> 5;
    if (lane == 0) { sh1[wave] = s1; sh2[wave] = s2; }
    __syncthreads();
    if (threadIdx.x == 0) {
        float a = 0.f, c = 0.f;
        for (int w = 0; w < 8; ++w) { a += sh1[w]; c += sh2[w]; }
        float mean = a / D;
        float var  = c / D - mean * mean;
        stats[0] = mean;
        stats[1] = rsqrtf(var + 1e-5f);
    }
    __syncthreads();
    const float mean = stats[0], inv = stats[1];
    __bf16* orow = out + row * D;
    for (int j = threadIdx.x; j < D; j += 256)
        orow[j] = (__bf16)((xr[j] - mean) * inv * g[j] + bb[j]);
}

__global__ void patchify_kernel(const float* __restrict__ img,
                                __bf16* __restrict__ Ap, long total)
{
    long idx = (long)blockIdx.x * blockDim.x + threadIdx.x;
    if (idx >= total) return;
    int d = (int)(idx % DIM);
    long t = idx / DIM;
    int p = (int)(t % NPATCH);
    int b = (int)(t / NPATCH);
    int c   = d >> 8;
    int rem = d & 255;
    int dy = rem >> 4, dx = rem & 15;
    int py = p / 14,  px = p % 14;
    long src = (((long)b * 3 + c) * 224 + (py * 16 + dy)) * 224 + (px * 16 + dx);
    Ap[idx] = (__bf16)img[src];
}

__global__ void assemble_kernel(const __bf16* __restrict__ tok,
                                const float* __restrict__ cls,
                                float* __restrict__ x, long total)
{
    long idx = (long)blockIdx.x * blockDim.x + threadIdx.x;
    if (idx >= total) return;
    int d = (int)(idx % DIM);
    long t = idx / DIM;
    int s = (int)(t % SEQ);
    int b = (int)(t / SEQ);
    float val = (s == 0) ? cls[d]
                         : (float)tok[((long)b * NPATCH + (s - 1)) * DIM + d];
    int jeven = d & ~1;
    float ang = (float)s * __expf(-(float)jeven * (9.2103403719761836f / 768.f));
    val += (d & 1) ? __cosf(ang) : __sinf(ang);
    x[idx] = val;
}

__global__ void cls_extract_kernel(const float* __restrict__ x,
                                   __bf16* __restrict__ out)
{
    int idx = blockIdx.x * 256 + threadIdx.x;
    if (idx >= BATCH * DIM) return;
    int b = idx / DIM, d = idx % DIM;
    out[idx] = (__bf16)x[(long)b * SEQ * DIM + d];
}

__global__ __launch_bounds__(256) void softmax_kernel(
    const float* __restrict__ logits, float* __restrict__ out, int N)
{
    const int row = blockIdx.x;
    const float* lr = logits + (long)row * N;
    float mx = -1e30f;
    for (int j = threadIdx.x; j < N; j += 256) mx = fmaxf(mx, lr[j]);
    for (int m = 16; m >= 1; m >>= 1) mx = fmaxf(mx, __shfl_xor(mx, m, 32));
    __shared__ float sh[8], stat[2];
    const int lane = threadIdx.x & 31, wave = threadIdx.x >> 5;
    if (lane == 0) sh[wave] = mx;
    __syncthreads();
    if (threadIdx.x == 0) {
        float v = -1e30f;
        for (int w = 0; w < 8; ++w) v = fmaxf(v, sh[w]);
        stat[0] = v;
    }
    __syncthreads();
    mx = stat[0];
    float sum = 0.f;
    for (int j = threadIdx.x; j < N; j += 256) sum += __expf(lr[j] - mx);
    for (int m = 16; m >= 1; m >>= 1) sum += __shfl_xor(sum, m, 32);
    if (lane == 0) sh[wave] = sum;
    __syncthreads();
    if (threadIdx.x == 0) {
        float v = 0.f;
        for (int w = 0; w < 8; ++w) v += sh[w];
        stat[1] = v;
    }
    __syncthreads();
    const float inv = 1.f / stat[1];
    for (int j = threadIdx.x; j < N; j += 256)
        out[(long)row * N + j] = __expf(lr[j] - mx) * inv;
}

// ============================================================================
// Host-side dispatch
// ============================================================================
static inline size_t swz_elems(int K, int N64, int Z) {
    return (size_t)Z * (size_t)(K / 32) * (size_t)(N64 / 16) * 512;
}

static void launch_swz(hipStream_t st, const float* Win, __bf16* Wout,
                       int K, int N, int N64, int Z, long inStrideZ)
{
    long total = (long)swz_elems(K, N64, Z);
    weight_swizzle_kernel<<<(unsigned)((total + 255) / 256), 256, 0, st>>>(
        Win, Wout, K, N, N64, inStrideZ, total);
}

template <int ACT, int HAS_RES, int OUT_BF16>
static void launch_gemm(hipStream_t st,
                        const __bf16* A, int lda, long aOffZ,
                        const __bf16* Wsw, long wOffZ, int nBlocks,
                        const float* bias, long bOffZ,
                        void* C, int ldc, long cOffZ,
                        const float* Res, int M, int N, int K, int Z)
{
    dim3 grid((N + 63) / 64, (M + 127) / 128, Z);
    gemm_wmma_kernel<ACT, HAS_RES, OUT_BF16><<<grid, 256, 0, st>>>(
        A, lda, aOffZ, Wsw, wOffZ, nBlocks, bias, bOffZ, C, ldc, cOffZ,
        Res, M, N, K);
}

extern "C" void kernel_launch(void* const* d_in, const int* in_sizes, int n_in,
                              void* d_out, int out_size, void* d_ws, size_t ws_size,
                              hipStream_t stream)
{
    (void)in_sizes; (void)n_in; (void)out_size; (void)ws_size;
    const float* images = (const float*)d_in[0];
    const float* Wp   = (const float*)d_in[1];
    const float* bp   = (const float*)d_in[2];
    const float* cls  = (const float*)d_in[3];
    const float* ln1g = (const float*)d_in[4];
    const float* ln1b = (const float*)d_in[5];
    const float* Wq   = (const float*)d_in[6];
    const float* bq   = (const float*)d_in[7];
    const float* Wk   = (const float*)d_in[8];
    const float* bk   = (const float*)d_in[9];
    const float* Wv   = (const float*)d_in[10];
    const float* bv   = (const float*)d_in[11];
    const float* Wpr  = (const float*)d_in[12];
    const float* bpr  = (const float*)d_in[13];
    const float* Wo   = (const float*)d_in[14];
    const float* bo   = (const float*)d_in[15];
    const float* ln2g = (const float*)d_in[16];
    const float* ln2b = (const float*)d_in[17];
    const float* W1   = (const float*)d_in[18];
    const float* b1   = (const float*)d_in[19];
    const float* W2   = (const float*)d_in[20];
    const float* b2   = (const float*)d_in[21];
    const float* Wh   = (const float*)d_in[22];
    const float* bh   = (const float*)d_in[23];

    // ---- workspace carve (activations first; weights last so padded-tile
    //      over-reads always land inside the workspace) ----
    char* p = (char*)d_ws;
    auto alloc = [&](size_t bytes) -> char* {
        char* r = p;
        p += (bytes + 255) & ~(size_t)255;
        return r;
    };
    const long   BSD = (long)M_TOK * DIM;
    float*  x   = (float*)alloc((size_t)BSD * 4);
    __bf16* hb  = (__bf16*)alloc((size_t)BSD * 2);
    __bf16* qb  = (__bf16*)alloc((size_t)BSD * 2);
    __bf16* kb  = (__bf16*)alloc((size_t)BSD * 2);
    __bf16* vb  = (__bf16*)alloc((size_t)BSD * 2);
    __bf16* ob  = (__bf16*)alloc((size_t)BSD * 2);
    __bf16* ff  = (__bf16*)alloc((size_t)M_TOK * FFDIM * 2);
    __bf16* patchA = ff;   // alias: used only before the layer loop
    __bf16* clsA   = (__bf16*)alloc((size_t)BATCH * DIM * 2);
    float*  logits = (float*)alloc((size_t)BATCH * NCLS * 4);
    __bf16* WpS  = (__bf16*)alloc(swz_elems(DIM, DIM, 1) * 2);
    __bf16* WqS  = (__bf16*)alloc(swz_elems(HDIM, HDIM, LAYERS * HEADS) * 2);
    __bf16* WkS  = (__bf16*)alloc(swz_elems(HDIM, HDIM, LAYERS * HEADS) * 2);
    __bf16* WvS  = (__bf16*)alloc(swz_elems(HDIM, HDIM, LAYERS * HEADS) * 2);
    __bf16* WprS = (__bf16*)alloc(swz_elems(HDIM, HDIM, LAYERS * HEADS) * 2);
    __bf16* WoS  = (__bf16*)alloc(swz_elems(DIM, DIM, LAYERS) * 2);
    __bf16* W1S  = (__bf16*)alloc(swz_elems(DIM, FFDIM, LAYERS) * 2);
    __bf16* W2S  = (__bf16*)alloc(swz_elems(FFDIM, DIM, LAYERS) * 2);
    __bf16* WhS  = (__bf16*)alloc(swz_elems(DIM, 1024, 1) * 2);

    // ---- weight pre-swizzle (f32 -> bf16 fragment blocks) ----
    launch_swz(stream, Wp,  WpS,  DIM,  DIM,  DIM,  1, 0);
    launch_swz(stream, Wq,  WqS,  HDIM, HDIM, HDIM, LAYERS * HEADS, HDIM * HDIM);
    launch_swz(stream, Wk,  WkS,  HDIM, HDIM, HDIM, LAYERS * HEADS, HDIM * HDIM);
    launch_swz(stream, Wv,  WvS,  HDIM, HDIM, HDIM, LAYERS * HEADS, HDIM * HDIM);
    launch_swz(stream, Wpr, WprS, HDIM, HDIM, HDIM, LAYERS * HEADS, HDIM * HDIM);
    launch_swz(stream, Wo,  WoS,  DIM,  DIM,  DIM,  LAYERS, (long)DIM * DIM);
    launch_swz(stream, W1,  W1S,  DIM,  FFDIM, FFDIM, LAYERS, (long)DIM * FFDIM);
    launch_swz(stream, W2,  W2S,  FFDIM, DIM,  DIM,  LAYERS, (long)FFDIM * DIM);
    launch_swz(stream, Wh,  WhS,  DIM,  NCLS, 1024, 1, 0);

    const size_t headWz = swz_elems(HDIM, HDIM, HEADS);        // per-layer, per-tensor
    const long   headZ  = (long)swz_elems(HDIM, HDIM, 1);      // per-head = 4096
    const size_t WoLz   = swz_elems(DIM, DIM, 1);
    const size_t W1Lz   = swz_elems(DIM, FFDIM, 1);
    const size_t W2Lz   = swz_elems(FFDIM, DIM, 1);

    // ---- patch embed ----
    {
        long total = (long)BATCH * NPATCH * DIM;
        patchify_kernel<<<(unsigned)((total + 255) / 256), 256, 0, stream>>>(
            images, patchA, total);
    }
    launch_gemm<0, 0, 1>(stream, patchA, DIM, 0, WpS, 0, DIM / 16, bp, 0,
                         hb, DIM, 0, nullptr, BATCH * NPATCH, DIM, DIM, 1);
    assemble_kernel<<<(unsigned)((BSD + 255) / 256), 256, 0, stream>>>(hb, cls, x, BSD);

    const int M = M_TOK;
    for (int l = 0; l < LAYERS; ++l) {
        const long bz = (long)l * HEADS * HDIM;

        layernorm_kernel<<<M, 256, 0, stream>>>(x, ln1g + (long)l * DIM,
                                                ln1b + (long)l * DIM, hb, DIM);
        launch_gemm<0, 0, 1>(stream, hb, DIM, HDIM, WqS + (size_t)l * headWz, headZ, 4,
                             bq + bz, HDIM, qb, DIM, HDIM, nullptr, M, HDIM, HDIM, HEADS);
        launch_gemm<0, 0, 1>(stream, hb, DIM, HDIM, WkS + (size_t)l * headWz, headZ, 4,
                             bk + bz, HDIM, kb, DIM, HDIM, nullptr, M, HDIM, HDIM, HEADS);
        launch_gemm<0, 0, 1>(stream, hb, DIM, HDIM, WvS + (size_t)l * headWz, headZ, 4,
                             bv + bz, HDIM, vb, DIM, HDIM, nullptr, M, HDIM, HDIM, HEADS);

        attention_kernel<<<BATCH * HEADS, 256, 0, stream>>>(qb, kb, vb, ob);

        launch_gemm<0, 0, 1>(stream, ob, DIM, HDIM, WprS + (size_t)l * headWz, headZ, 4,
                             bpr + bz, HDIM, hb, DIM, HDIM, nullptr, M, HDIM, HDIM, HEADS);
        launch_gemm<0, 1, 0>(stream, hb, DIM, 0, WoS + (size_t)l * WoLz, 0, DIM / 16,
                             bo + (long)l * DIM, 0, x, DIM, 0, x, M, DIM, DIM, 1);

        layernorm_kernel<<<M, 256, 0, stream>>>(x, ln2g + (long)l * DIM,
                                                ln2b + (long)l * DIM, hb, DIM);
        launch_gemm<1, 0, 1>(stream, hb, DIM, 0, W1S + (size_t)l * W1Lz, 0, FFDIM / 16,
                             b1 + (long)l * FFDIM, 0, ff, FFDIM, 0, nullptr, M, FFDIM, DIM, 1);
        launch_gemm<0, 1, 0>(stream, ff, FFDIM, 0, W2S + (size_t)l * W2Lz, 0, DIM / 16,
                             b2 + (long)l * DIM, 0, x, DIM, 0, x, M, DIM, FFDIM, 1);
    }

    // ---- classifier head on cls tokens ----
    cls_extract_kernel<<<(BATCH * DIM + 255) / 256, 256, 0, stream>>>(x, clsA);
    launch_gemm<0, 0, 0>(stream, clsA, DIM, 0, WhS, 0, 1024 / 16, bh, 0,
                         logits, NCLS, 0, nullptr, BATCH, NCLS, DIM, 1);
    softmax_kernel<<<BATCH, 256, 0, stream>>>(logits, (float*)d_out, NCLS);
}
